// Fold_45801531245057
// MI455X (gfx1250) — matrix operand order repaired and checked by reading, not compiled, not run on hardware
//
#include <hip/hip_runtime.h>
#include <hip/hip_bf16.h>
#include <stdint.h>

typedef __attribute__((ext_vector_type(16))) _Float16 v16h;
typedef __attribute__((ext_vector_type(8)))  float    v8f;

#define KEYS   (1u << 24)      // 256^3 parent keys (b == 0 in this workload)
#define TILE   2048u           // elements scanned per workgroup (8 waves x 256)
#define NTILES (KEYS / TILE)   // 8192

// Low 32 bits of a flat pointer to LDS == LDS byte offset (ISA 10.2 aperture map).
__device__ __forceinline__ uint32_t lds_addr32(const void* p) {
  return (uint32_t)(uintptr_t)p;
}

// ---------------------------------------------------------------------------
// Zero helper (harness poisons d_out/d_ws with 0xAA).
// ---------------------------------------------------------------------------
__global__ void zero_u4_kernel(uint4* __restrict__ p, uint32_t n4) {
  uint32_t i = blockIdx.x * blockDim.x + threadIdx.x;
  if (i < n4) p[i] = make_uint4(0u, 0u, 0u, 0u);
}

// ---------------------------------------------------------------------------
// Pass 1: mark presence of each parent key (plain store of 1; races benign).
// ---------------------------------------------------------------------------
__global__ void mark_keys_kernel(const int4* __restrict__ coords,
                                 uint32_t* __restrict__ flags, int n) {
  int i = blockIdx.x * blockDim.x + threadIdx.x;
  if (i >= n) return;
  __builtin_prefetch(coords + i + 4096, 0, 0);   // -> global_prefetch_b8
  int4 c = coords[i];                             // [b, x, y, z]
  uint32_t key = ((uint32_t)c.x << 24) |
                 ((uint32_t)(c.y >> 1) << 16) |
                 ((uint32_t)(c.z >> 1) << 8)  |
                  (uint32_t)(c.w >> 1);
  flags[key] = 1u;
}

// ---------------------------------------------------------------------------
// Pass 2a: per-tile exclusive scan of 2048 flags using WMMA.
// One wave scans 256 elements viewed as a 16x16 matrix X:
//   T = X * U          (U upper-triangular ones)  -> row-inclusive prefixes
//   S = Ls * R + T     (Ls strict-lower ones, R = rowtot broadcast) -> full scan
// f16 inputs are 0/1 and row totals <= 256, f32 accumulate => exact integers.
// The 1KB chunk is staged global->LDS with the CDNA5 async DMA path
// (GLOBAL_LOAD_ASYNC_TO_LDS_B128, ASYNCcnt) and consumed twice from LDS.
// ISA layouts (CDNA5 ISA 7.12.2):
//   A(16x32 f16): lane L -> row m=L&15, K-half h=L>>4; elems e=0..7 are K=8h+e.
//   B(32x16 f16): lane L -> col N=L&15, K-half h=L>>4; elems e=0..7 are K=8h+e.
//   C/D(16x16 f32): lane L -> col n=L&15; vgpr v -> row m=v+8*(L>>4).
// ---------------------------------------------------------------------------
__global__ void scan_tiles_kernel(uint32_t* __restrict__ flags,
                                  uint32_t* __restrict__ blockSums) {
  __shared__ uint32_t stage[8][256];    // one 1KB chunk per wave
  __shared__ float    rowtot[8][16];
  __shared__ float    wsum[8];

  const int lane = threadIdx.x & 31;
  const int wave = threadIdx.x >> 5;        // 0..7
  const int ln   = lane & 15;
  const int h    = lane >> 4;
  const uint32_t chunk = blockIdx.x * TILE + (uint32_t)wave * 256u;

  // --- async stage: 2 x (32 lanes x 16B) = 1KB into this wave's LDS slot ---
  const uint32_t* gsrc  = flags + chunk + lane * 4;     // 4 dwords per lane
  uint32_t ldsa = lds_addr32(&stage[wave][lane * 4]);
  asm volatile("global_load_async_to_lds_b128 %0, %1, off"
               :: "v"(ldsa), "v"(gsrc) : "memory");
  asm volatile("global_load_async_to_lds_b128 %0, %1, off offset:512"
               :: "v"(ldsa), "v"(gsrc) : "memory");
  asm volatile("s_wait_asynccnt 0x0" ::: "memory");     // wave-local staging

  // Load A1 = X zero-padded to 16x32: lane holds X[ln][8h .. 8h+7] (from LDS).
  const uint4* sp = reinterpret_cast<const uint4*>(&stage[wave][ln * 16 + h * 8]);
  uint4 q0 = sp[0];
  uint4 q1 = sp[1];
  float xa[8] = { (float)q0.x, (float)q0.y, (float)q0.z, (float)q0.w,
                  (float)q1.x, (float)q1.y, (float)q1.z, (float)q1.w };

  v16h A1 = {};
#pragma unroll
  for (int e = 0; e < 8; ++e) A1[e] = (_Float16)xa[e];

  // B1 = U (16x16 upper-triangular ones) zero-padded to 32x16; col N=ln, elem e ~ K=8h+e.
  v16h B1 = {};
#pragma unroll
  for (int e = 0; e < 8; ++e)
    B1[e] = (_Float16)(((8 * h + e) <= ln) ? 1.0f : 0.0f);

  v8f zero_c = {};
  v8f T = __builtin_amdgcn_wmma_f32_16x16x32_f16(false, A1, false, B1,
                                                 (short)0, zero_c, false, false);

  // Row totals = column 15 of T (lives in lanes 15 and 31), via LDS transpose.
  if (lane == 15) {
#pragma unroll
    for (int v = 0; v < 8; ++v) rowtot[wave][v] = T[v];
  }
  if (lane == 31) {
#pragma unroll
    for (int v = 0; v < 8; ++v) rowtot[wave][8 + v] = T[v];
  }
  __syncthreads();

  // A2 = strict-lower-triangular ones zero-padded: elem e ~ K=8h+e, row ln.
  v16h A2 = {};
#pragma unroll
  for (int e = 0; e < 8; ++e)
    A2[e] = (_Float16)(((8 * h + e) < ln) ? 1.0f : 0.0f);

  // B2: elem e ~ K=8h+e holds rowtot[K] (K 16..31 zero), any column.
  v16h B2 = {};
#pragma unroll
  for (int e = 0; e < 8; ++e) B2[e] = (_Float16)rowtot[wave][8 * h + e];

  v8f S = __builtin_amdgcn_wmma_f32_16x16x32_f16(false, A2, false, B2,
                                                 (short)0, T, false, false);

  // Wave total = S(15,15) = vgpr7 of lane 31; combine waves within the tile.
  float wtot = __shfl(S[7], 31, 32);
  if (lane == 0) wsum[wave] = wtot;
  __syncthreads();

  float woff = 0.0f;
#pragma unroll
  for (int w = 0; w < 8; ++w) woff += (w < wave) ? wsum[w] : 0.0f;

  // Write back tile-local EXCLUSIVE scan: lane owns S rows v+8h, col ln.
  // Originals come from the LDS stage (no second global read).
#pragma unroll
  for (int v = 0; v < 8; ++v) {
    uint32_t idx  = (uint32_t)(v + 8 * h) * 16u + (uint32_t)ln;
    uint32_t orig = stage[wave][idx];
    flags[chunk + idx] = (uint32_t)(S[v] + woff) - orig;
  }

  if (threadIdx.x == 0) {
    float bt = 0.0f;
#pragma unroll
    for (int w = 0; w < 8; ++w) bt += wsum[w];
    blockSums[blockIdx.x] = (uint32_t)bt;       // tile total
  }
}

// ---------------------------------------------------------------------------
// Pass 2b: exclusive scan of the 8192 tile sums (single workgroup).
// ---------------------------------------------------------------------------
__global__ void scan_level2_kernel(uint32_t* __restrict__ bs,
                                   uint32_t* __restrict__ total, int n) {
  __shared__ uint32_t sh[256];
  __shared__ uint32_t carry_sh;
  const int t = threadIdx.x;
  if (t == 0) carry_sh = 0u;
  __syncthreads();
  for (int base = 0; base < n; base += 256) {
    uint32_t v = bs[base + t];
    sh[t] = v;
    __syncthreads();
    for (int off = 1; off < 256; off <<= 1) {
      uint32_t u = (t >= off) ? sh[t - off] : 0u;
      __syncthreads();
      sh[t] += u;
      __syncthreads();
    }
    uint32_t incl = sh[t];
    uint32_t excl = carry_sh + incl - v;
    __syncthreads();
    if (t == 255) carry_sh += incl;
    bs[base + t] = excl;
    __syncthreads();
  }
  if (t == 0) total[0] = carry_sh;              // number of unique keys
}

// ---------------------------------------------------------------------------
// Pass 2c: add tile offsets -> flags[] becomes the global rank table.
// ---------------------------------------------------------------------------
__global__ void add_offsets_kernel(uint32_t* __restrict__ flags,
                                   const uint32_t* __restrict__ bs) {
  uint32_t off  = bs[blockIdx.x];
  uint32_t base = blockIdx.x * TILE + threadIdx.x * 8u;
  uint4* p = reinterpret_cast<uint4*>(flags + base);
  uint4 a = p[0], b = p[1];
  a.x += off; a.y += off; a.z += off; a.w += off;
  b.x += off; b.y += off; b.z += off; b.w += off;
  p[0] = a; p[1] = b;
}

// ---------------------------------------------------------------------------
// Pass 3: scatter-add features at rank (global_atomic_add_f32, no return).
// ---------------------------------------------------------------------------
__global__ void scatter_feats_kernel(const int4* __restrict__ coords,
                                     const float* __restrict__ feats,
                                     const uint32_t* __restrict__ ranks,
                                     float* __restrict__ out_feats, int n) {
  int i = blockIdx.x * blockDim.x + threadIdx.x;
  if (i >= n) return;
  __builtin_prefetch(coords + i + 4096, 0, 0);
  int4 c = coords[i];
  uint32_t key = ((uint32_t)c.x << 24) |
                 ((uint32_t)(c.y >> 1) << 16) |
                 ((uint32_t)(c.z >> 1) << 8)  |
                  (uint32_t)(c.w >> 1);
  int off8 = ((c.y & 1) << 2) | ((c.z & 1) << 1) | (c.w & 1);
  uint32_t idx = ranks[key];
  atomicAdd(out_feats + (size_t)idx * 8 + off8, feats[i]);
}

// ---------------------------------------------------------------------------
// Pass 4: decode present keys (ascending key order == sorted unique order).
// Neighbor rank comes via LDS so the 64MB rank table is read once, not twice.
// ---------------------------------------------------------------------------
__global__ void write_coords_kernel(const uint32_t* __restrict__ ranks,
                                    const uint32_t* __restrict__ total,
                                    float4* __restrict__ out_coords) {
  __shared__ uint32_t sr[257];
  const int t = threadIdx.x;
  uint32_t k = blockIdx.x * blockDim.x + t;
  sr[t] = ranks[k];
  if (t == 255) sr[256] = (k + 1u < KEYS) ? ranks[k + 1u] : total[0];
  __syncthreads();
  uint32_t r  = sr[t];
  uint32_t rn = sr[t + 1];
  if (rn != r) {                                 // key k is present, rank r
    float4 oc;
    oc.x = (float)(k >> 24);                     // b
    oc.y = (float)((k >> 16) & 255u);            // x
    oc.z = (float)((k >> 8) & 255u);             // y
    oc.w = (float)(k & 255u);                    // z
    out_coords[r] = oc;
  }
}

__global__ void fill_invalid_kernel(float4* __restrict__ out_coords,
                                    const uint32_t* __restrict__ total, int n) {
  uint32_t i = blockIdx.x * blockDim.x + threadIdx.x;
  if (i < (uint32_t)n && i >= total[0])
    out_coords[i] = make_float4(-1.0f, -1.0f, -1.0f, -1.0f);
}

// ---------------------------------------------------------------------------
extern "C" void kernel_launch(void* const* d_in, const int* in_sizes, int n_in,
                              void* d_out, int out_size, void* d_ws, size_t ws_size,
                              hipStream_t stream) {
  const int4*  coords = (const int4*)d_in[0];
  const float* feats  = (const float*)d_in[1];
  const int n = in_sizes[0] / 4;                 // N points

  uint32_t* flags = (uint32_t*)d_ws;             // KEYS u32  (64 MB, L2-resident)
  uint32_t* bsums = flags + KEYS;                // NTILES u32
  uint32_t* total = bsums + NTILES;              // 1 u32

  float* out_coords = (float*)d_out;             // N*4 floats (values exact ints)
  float* out_feats  = out_coords + (size_t)n * 4; // N*8 floats

  const int B = 256;

  // zero rank table + out_feats accumulator
  zero_u4_kernel<<<(KEYS / 4 + B - 1) / B, B, 0, stream>>>((uint4*)flags, KEYS / 4);
  uint32_t nf4 = (uint32_t)n * 2u;               // n*8 floats / 4
  zero_u4_kernel<<<(nf4 + B - 1) / B, B, 0, stream>>>((uint4*)out_feats, nf4);

  mark_keys_kernel<<<(n + B - 1) / B, B, 0, stream>>>(coords, flags, n);

  scan_tiles_kernel<<<NTILES, B, 0, stream>>>(flags, bsums);
  scan_level2_kernel<<<1, B, 0, stream>>>(bsums, total, NTILES);
  add_offsets_kernel<<<NTILES, B, 0, stream>>>(flags, bsums);

  scatter_feats_kernel<<<(n + B - 1) / B, B, 0, stream>>>(coords, feats, flags,
                                                          out_feats, n);
  write_coords_kernel<<<KEYS / B, B, 0, stream>>>(flags, total,
                                                  (float4*)out_coords);
  fill_invalid_kernel<<<(n + B - 1) / B, B, 0, stream>>>((float4*)out_coords,
                                                         total, n);
}
